// TextSelfAttention_53085795778840
// MI455X (gfx1250) — matrix operand order repaired and checked by reading, not compiled, not run on hardware
//
#include <hip/hip_runtime.h>

typedef __attribute__((ext_vector_type(16))) __bf16       v16bf;
typedef __attribute__((ext_vector_type(8)))  float        v8f;
typedef __attribute__((ext_vector_type(4)))  float        v4f;
typedef __attribute__((ext_vector_type(4)))  unsigned int v4u;

union FragU { v4u u[2]; v16bf v; };

#define DIM    1024
#define SEQ    2048
#define NBATCH 8

__device__ __forceinline__ v8f wmma_bf16(v16bf a, v16bf b, v8f c) {
  // (neg_a, A, neg_b, B, c_mod, C, reuse_a, reuse_b)
  return __builtin_amdgcn_wmma_f32_16x16x32_bf16(false, a, false, b, (short)0, c,
                                                 false, false);
}

// ---------------------------------------------------------------------------
// Kernel 0: bulk f32 -> bf16 conversion. Each thread converts 8 elements.
// ---------------------------------------------------------------------------
__global__ __launch_bounds__(256) void cvt_bf16_kernel(
    const float* __restrict__ in, __bf16* __restrict__ out)
{
  const size_t i = (size_t)blockIdx.x * blockDim.x + threadIdx.x;
  v4f a = ((const v4f*)in)[2 * i];
  v4f b = ((const v4f*)in)[2 * i + 1];
  union { __bf16 h[8]; v4u u; } t;
#pragma unroll
  for (int j = 0; j < 4; ++j) {
    t.h[j]     = (__bf16)a[j];
    t.h[4 + j] = (__bf16)b[j];
  }
  ((v4u*)out)[i] = t.u;
}

// ---------------------------------------------------------------------------
// Kernel 1: y = x @ W^T + b  (bf16 in, bf16 out).
//   transpose_out == 0 : out[s][e]    -- Q, K
//   transpose_out == 1 : out[b][e][s] -- V (transposed for the PV stage)
// Grid: (DIM/256, (NBATCH*SEQ)/16), block 256 (8 waves).
// Wave tile = 16s x 32e (2 accumulators): A fragment reused for 2 WMMAs.
// The 16 x-rows are shared by all 8 waves -> staged once in LDS.
// ---------------------------------------------------------------------------
__global__ __launch_bounds__(256) void qkv_proj_kernel(
    const __bf16* __restrict__ x, const __bf16* __restrict__ W,
    const float* __restrict__ bias, __bf16* __restrict__ out, int transpose_out)
{
  const int lane = threadIdx.x & 31;
  const int wave = threadIdx.x >> 5;
  const int hf   = lane >> 4;
  const int ln   = lane & 15;
  const int e0   = blockIdx.x * 256 + wave * 32;
  const int s0   = blockIdx.y * 16;

  __shared__ __align__(16) __bf16 sX[16 * DIM];  // 32 KB
  {
    const v4u* src = (const v4u*)(x + (size_t)s0 * DIM);  // 2048 x 16B
    v4u* dst = (v4u*)sX;
#pragma unroll
    for (int i = 0; i < 8; ++i) dst[threadIdx.x + 256 * i] = src[threadIdx.x + 256 * i];
  }
  __syncthreads();

  const char* arow  = (const char*)sX + ln * (DIM * 2);                 // A row
  const char* wrow0 = (const char*)(W + (size_t)(e0 + ln) * DIM);       // B^T rows
  const char* wrow1 = (const char*)(W + (size_t)(e0 + 16 + ln) * DIM);

  v8f acc0 = {}, acc1 = {};
#pragma unroll 4
  for (int d0 = 0; d0 < DIM; d0 += 32) {
    FragU a, b0, b1;
    a.u[0]  = *(const v4u*)(arow + d0 * 2 + hf * 16);       // K = hf*8    + e
    a.u[1]  = *(const v4u*)(arow + d0 * 2 + 32 + hf * 16);  // K = 16+hf*8 + e
    b0.u[0] = *(const v4u*)(wrow0 + d0 * 2 + hf * 16);      // K = hf*8    + e
    b0.u[1] = *(const v4u*)(wrow0 + d0 * 2 + 32 + hf * 16);
    b1.u[0] = *(const v4u*)(wrow1 + d0 * 2 + hf * 16);
    b1.u[1] = *(const v4u*)(wrow1 + d0 * 2 + 32 + hf * 16);
    acc0 = wmma_bf16(a.v, b0.v, acc0);
    acc1 = wmma_bf16(a.v, b1.v, acc1);
  }

  const float bc0 = bias[e0 + ln];
  const float bc1 = bias[e0 + 16 + ln];
  if (!transpose_out) {
    // C layout: lane holds column n=ln, vgpr v holds row m = v + 8*hf
#pragma unroll
    for (int v = 0; v < 8; ++v) {
      const int m = v + 8 * hf;
      out[(size_t)(s0 + m) * DIM + e0 + ln]      = (__bf16)(acc0[v] + bc0);
      out[(size_t)(s0 + m) * DIM + e0 + 16 + ln] = (__bf16)(acc1[v] + bc1);
    }
  } else {
    // vT[b][e][s]: rows m are consecutive s -> one 16B store per lane per tile
    const int b  = s0 >> 11;  // SEQ = 2048
    const int sl = s0 & (SEQ - 1);
    union { __bf16 h[8]; v4u u; } t0, t1;
#pragma unroll
    for (int v = 0; v < 8; ++v) {
      t0.h[v] = (__bf16)(acc0[v] + bc0);
      t1.h[v] = (__bf16)(acc1[v] + bc1);
    }
    *(v4u*)(out + ((size_t)b * DIM + e0 + ln) * SEQ + sl + 8 * hf)      = t0.u;
    *(v4u*)(out + ((size_t)b * DIM + e0 + 16 + ln) * SEQ + sl + 8 * hf) = t1.u;
  }
}

// ---------------------------------------------------------------------------
// Kernel 2: flash attention. One workgroup per (batch, 16-query tile).
// Q tile staged in LDS. Key block = 256 keys: each wave computes two 16x16
// score subtiles (Q fragment reused for 2 WMMAs), shared online softmax via
// LDS, then each wave accumulates a 128-wide d-slice of O with P@V WMMAs
// (P fragment reused for 8 WMMAs; V^T fragments from global, L2-resident).
// ---------------------------------------------------------------------------
__global__ __launch_bounds__(256) void flash_attn_kernel(
    const __bf16* __restrict__ q, const __bf16* __restrict__ k,
    const __bf16* __restrict__ vT, float* __restrict__ out)
{
  const int lane = threadIdx.x & 31;
  const int wave = threadIdx.x >> 5;
  const int hf   = lane >> 4;
  const int ln   = lane & 15;
  const int b    = blockIdx.x;
  const int q0   = blockIdx.y * 16;

  const __bf16* qb = q  + ((size_t)b * SEQ + q0) * DIM;
  const __bf16* kb = k  + (size_t)b * SEQ * DIM;
  const __bf16* vb = vT + (size_t)b * DIM * SEQ;

  __shared__ __align__(16) __bf16 sQ[16 * DIM];  // 32 KB, reused for all blocks
  __shared__ __align__(16) __bf16 sP[16 * 256];  // 8 KB, row-major [q][key]
  __shared__ float sPartMax[8][16];
  __shared__ float sPartSum[8][16];
  __shared__ float sM[16], sL[16], sAlpha[16], sMnew[16];

  {
    const v4u* src = (const v4u*)qb;
    v4u* dst = (v4u*)sQ;
#pragma unroll
    for (int i = 0; i < 8; ++i) dst[threadIdx.x + 256 * i] = src[threadIdx.x + 256 * i];
  }
  if (threadIdx.x < 16) { sM[threadIdx.x] = -1e30f; sL[threadIdx.x] = 0.0f; }
  __syncthreads();

  v8f O[8];
#pragma unroll
  for (int s = 0; s < 8; ++s) O[s] = (v8f){};

  const char* qrow = (const char*)sQ + ln * (DIM * 2);

  for (int k0 = 0; k0 < SEQ; k0 += 256) {
    // prefetch next key block's rows for this wave into cache
    if (k0 + 256 < SEQ) {
      __builtin_prefetch(kb + (size_t)(k0 + 256 + wave * 32 + ln) * DIM, 0, 1);
      __builtin_prefetch(kb + (size_t)(k0 + 256 + wave * 32 + 16 + ln) * DIM, 0, 1);
    }

    // ---- 1) scores: 16 queries x 32 keys [k0+wave*32, +32), 2 subtiles ----
    v8f sc0 = {}, sc1 = {};
    const char* krow0 = (const char*)(kb + (size_t)(k0 + wave * 32 + ln) * DIM);
    const char* krow1 = (const char*)(kb + (size_t)(k0 + wave * 32 + 16 + ln) * DIM);
#pragma unroll 4
    for (int d0 = 0; d0 < DIM; d0 += 32) {
      FragU a, b0, b1;
      a.u[0]  = *(const v4u*)(qrow + d0 * 2 + hf * 16);
      a.u[1]  = *(const v4u*)(qrow + d0 * 2 + 32 + hf * 16);
      b0.u[0] = *(const v4u*)(krow0 + d0 * 2 + hf * 16);
      b0.u[1] = *(const v4u*)(krow0 + d0 * 2 + 32 + hf * 16);
      b1.u[0] = *(const v4u*)(krow1 + d0 * 2 + hf * 16);
      b1.u[1] = *(const v4u*)(krow1 + d0 * 2 + 32 + hf * 16);
      sc0 = wmma_bf16(a.v, b0.v, sc0);
      sc1 = wmma_bf16(a.v, b1.v, sc1);
    }
    float s0v[8], s1v[8];
#pragma unroll
    for (int v = 0; v < 8; ++v) {
      s0v[v] = sc0[v] * 0.03125f;  // 1/sqrt(1024)
      s1v[v] = sc1[v] * 0.03125f;
    }

    // ---- 2) per-row max within wave ----
    float mx[8];
#pragma unroll
    for (int v = 0; v < 8; ++v) mx[v] = fmaxf(s0v[v], s1v[v]);
#pragma unroll
    for (int msk = 1; msk < 16; msk <<= 1)
#pragma unroll
      for (int v = 0; v < 8; ++v) mx[v] = fmaxf(mx[v], __shfl_xor(mx[v], msk, 32));
    if (ln == 0) {
#pragma unroll
      for (int v = 0; v < 8; ++v) sPartMax[wave][hf * 8 + v] = mx[v];
    }
    __syncthreads();

    // ---- 3) combine maxima, running m / alpha ----
    if (threadIdx.x < 16) {
      const int r = threadIdx.x;
      float mp = sPartMax[0][r];
#pragma unroll
      for (int w = 1; w < 8; ++w) mp = fmaxf(mp, sPartMax[w][r]);
      const float mold = sM[r];
      const float mnew = fmaxf(mold, mp);
      sMnew[r]  = mnew;
      sAlpha[r] = __expf(mold - mnew);
      sM[r]     = mnew;
    }
    __syncthreads();

    // ---- 4) p = exp(s - mnew); write P to LDS; row sums ----
    float sum8[8];
#pragma unroll
    for (int v = 0; v < 8; ++v) {
      const int m = v + 8 * hf;
      const float mn = sMnew[m];
      const float p0 = __expf(s0v[v] - mn);
      const float p1 = __expf(s1v[v] - mn);
      sP[m * 256 + wave * 32 + ln]      = (__bf16)p0;
      sP[m * 256 + wave * 32 + 16 + ln] = (__bf16)p1;
      sum8[v] = p0 + p1;
    }
#pragma unroll
    for (int msk = 1; msk < 16; msk <<= 1)
#pragma unroll
      for (int v = 0; v < 8; ++v) sum8[v] += __shfl_xor(sum8[v], msk, 32);
    if (ln == 0) {
#pragma unroll
      for (int v = 0; v < 8; ++v) sPartSum[wave][hf * 8 + v] = sum8[v];
    }
    __syncthreads();
    if (threadIdx.x < 16) {
      const int r = threadIdx.x;
      float s = 0.0f;
#pragma unroll
      for (int w = 0; w < 8; ++w) s += sPartSum[w][r];
      sL[r] = sL[r] * sAlpha[r] + s;
    }

    // ---- 5) O = O*alpha + P @ V  (this wave owns d-slice [wave*128, +128)) --
    float al[8];
#pragma unroll
    for (int v = 0; v < 8; ++v) al[v] = sAlpha[v + 8 * hf];
#pragma unroll
    for (int s = 0; s < 8; ++s)
#pragma unroll
      for (int v = 0; v < 8; ++v) O[s][v] *= al[v];

#pragma unroll
    for (int jc = 0; jc < 8; ++jc) {   // 32 keys per chunk, 256 per block
      FragU a;
      a.u[0] = *(const v4u*)((const char*)sP + ln * 512 + jc * 64 + hf * 16);
      a.u[1] = *(const v4u*)((const char*)sP + ln * 512 + jc * 64 + 32 + hf * 16);
#pragma unroll
      for (int s = 0; s < 8; ++s) {    // 8 d-subtiles of 16
        const char* vrow =
            (const char*)(vb + (size_t)(wave * 128 + s * 16 + ln) * SEQ);
        FragU bf;
        bf.u[0] = *(const v4u*)(vrow + (k0 + jc * 32) * 2 + hf * 16);
        bf.u[1] = *(const v4u*)(vrow + (k0 + jc * 32) * 2 + 32 + hf * 16);
        O[s] = wmma_bf16(a.v, bf.v, O[s]);
      }
    }
  }

  __syncthreads();
  float li[8];
#pragma unroll
  for (int v = 0; v < 8; ++v) li[v] = 1.0f / sL[v + 8 * hf];

#pragma unroll
  for (int s = 0; s < 8; ++s)
#pragma unroll
    for (int v = 0; v < 8; ++v) {
      const int m = v + 8 * hf;
      out[((size_t)b * SEQ + q0 + m) * DIM + wave * 128 + s * 16 + ln] =
          O[s][v] * li[v];
    }
}

// ---------------------------------------------------------------------------
extern "C" void kernel_launch(void* const* d_in, const int* in_sizes, int n_in,
                              void* d_out, int out_size, void* d_ws,
                              size_t ws_size, hipStream_t stream) {
  const float* x  = (const float*)d_in[0];
  const float* Wq = (const float*)d_in[1];
  const float* bq = (const float*)d_in[2];
  const float* Wk = (const float*)d_in[3];
  const float* bk = (const float*)d_in[4];
  const float* Wv = (const float*)d_in[5];
  const float* bv = (const float*)d_in[6];
  float* out = (float*)d_out;

  const size_t elems  = (size_t)NBATCH * SEQ * DIM;  // 16777216
  const size_t welems = (size_t)DIM * DIM;           // 1048576
  __bf16* qw  = (__bf16*)d_ws;
  __bf16* kw  = qw + elems;
  __bf16* vw  = kw + elems;   // stored transposed [b][d][s]
  __bf16* xw  = vw + elems;   // x in bf16
  __bf16* wqb = xw + elems;
  __bf16* wkb = wqb + welems;
  __bf16* wvb = wkb + welems;

  dim3 blk(256);
  // f32 -> bf16 preconversion (8 elements/thread)
  cvt_bf16_kernel<<<dim3(elems / (256 * 8)), blk, 0, stream>>>(x, xw);
  cvt_bf16_kernel<<<dim3(welems / (256 * 8)), blk, 0, stream>>>(Wq, wqb);
  cvt_bf16_kernel<<<dim3(welems / (256 * 8)), blk, 0, stream>>>(Wk, wkb);
  cvt_bf16_kernel<<<dim3(welems / (256 * 8)), blk, 0, stream>>>(Wv, wvb);

  dim3 g1(DIM / 256, (NBATCH * SEQ) / 16);
  qkv_proj_kernel<<<g1, blk, 0, stream>>>(xw, wqb, bq, qw, 0);
  qkv_proj_kernel<<<g1, blk, 0, stream>>>(xw, wkb, bk, kw, 0);
  qkv_proj_kernel<<<g1, blk, 0, stream>>>(xw, wvb, bv, vw, 1);

  dim3 g2(NBATCH, SEQ / 16);
  flash_attn_kernel<<<g2, blk, 0, stream>>>(qw, kw, vw, out);
}
